// GraphRewiringNetwork_29695403884554
// MI455X (gfx1250) — matrix operand order repaired and checked by reading, not compiled, not run on hardware
//
#include <hip/hip_runtime.h>
#include <hip/hip_bf16.h>

typedef __attribute__((ext_vector_type(16))) _Float16 v16h;
typedef __attribute__((ext_vector_type(8)))  float    v8f;

#define NUM_EDGES 800000
#define EMBED     128
#define IN_DIM    291
#define HID       145
#define KPAD      320
#define NPAD      160
#define NT        10     // N tiles of 16 (145 -> 160)
#define NC        9      // compute chunks: 0..7 = h_src/h_dst, 8 = scalar chunk (k-chunk 9)
#define THREADS   256
#define WAVES     8
#define GRID      800

__device__ __forceinline__ v16h pack16(float4 x0, float4 x1, float4 y0, float4 y1) {
    // elements 0..7 = first 8 floats (K = khalf..khalf+7)
    // elements 8..15 = second 8 floats (K = khalf+8..khalf+15, khalf = 16*half)
    v16h r;
    r[0]  = (_Float16)x0.x; r[1]  = (_Float16)x0.y; r[2]  = (_Float16)x0.z; r[3]  = (_Float16)x0.w;
    r[4]  = (_Float16)x1.x; r[5]  = (_Float16)x1.y; r[6]  = (_Float16)x1.z; r[7]  = (_Float16)x1.w;
    r[8]  = (_Float16)y0.x; r[9]  = (_Float16)y0.y; r[10] = (_Float16)y0.z; r[11] = (_Float16)y0.w;
    r[12] = (_Float16)y1.x; r[13] = (_Float16)y1.y; r[14] = (_Float16)y1.z; r[15] = (_Float16)y1.w;
    return r;
}

__global__ __launch_bounds__(THREADS) void edge_mlp_wmma(
    const float* __restrict__ h,
    const long long* __restrict__ ei,
    const float* __restrict__ cancer,
    const float* __restrict__ causal,
    const float* __restrict__ W1,
    const float* __restrict__ b1,
    const float* __restrict__ W2,
    const float* __restrict__ b2,
    float* __restrict__ out)
{
    // W1^T, padded, f16: sW[n*KPAD + k] = W1[k*HID + n]   (102,400 B)
    __shared__ _Float16 sW[NPAD * KPAD];
    // per-n bias with the constant cancer-chunk contribution folded in, and W2 column
    __shared__ float sBias[NPAD];
    __shared__ float sW2[NPAD];

    const int tid = threadIdx.x;

    for (int idx = tid; idx < NPAD * KPAD; idx += THREADS) {
        const int n = idx / KPAD;
        const int k = idx - n * KPAD;
        float v = 0.f;
        if (n < HID && k < IN_DIM) v = W1[k * HID + n];
        sW[idx] = (_Float16)v;
    }
    for (int n = tid; n < NPAD; n += THREADS) {
        float bias = 0.f, w2 = 0.f;
        if (n < HID) {
            float s = b1[n];
            #pragma unroll
            for (int k = 0; k < 32; ++k) s += cancer[k] * W1[(256 + k) * HID + n];
            bias = s;
            w2 = W2[n];
        }
        sBias[n] = bias;
        sW2[n] = w2;
    }
    __syncthreads();

    const int lane  = tid & 31;
    const int mloc  = lane & 15;   // row M within tile / column N within n-tile
    const int halfu = lane >> 4;   // 0: lanes 0-15, 1: lanes 16-31
    const float b2s = b2[0];

    const int wave = tid >> 5;
    const int ngroups = NUM_EDGES / 32;  // 25000, exact

    for (int g = blockIdx.x * WAVES + wave; g < ngroups; g += gridDim.x * WAVES) {
        const int ebase = g * 32;

        // Build A fragments for two 16-edge M-tiles, all 9 compute chunks (144 VGPRs).
        v16h A[2][NC];
        #pragma unroll
        for (int tile = 0; tile < 2; ++tile) {
            const int e = ebase + tile * 16 + mloc;
            const int s = (int)ei[e];
            const int d = (int)ei[NUM_EDGES + e];
            const float* hs = h + (long)s * EMBED;
            const float* hd = h + (long)d * EMBED;

            #pragma unroll
            for (int c = 0; c < 4; ++c) {
                const float* b0 = hs + c * 32 + halfu * 16;
                const float4* p0 = (const float4*)b0;
                const float4* p1 = (const float4*)(b0 + 8);
                A[tile][c] = pack16(p0[0], p0[1], p1[0], p1[1]);
            }
            #pragma unroll
            for (int c = 0; c < 4; ++c) {
                const float* b0 = hd + c * 32 + halfu * 16;
                const float4* p0 = (const float4*)b0;
                const float4* p1 = (const float4*)(b0 + 8);
                A[tile][4 + c] = pack16(p0[0], p0[1], p1[0], p1[1]);
            }

            const float ci = causal[s];
            const float cj = causal[d];
            v16h a9;
            #pragma unroll
            for (int j = 0; j < 16; ++j) a9[j] = (_Float16)0.f;
            if (halfu == 0) {        // K=288,289,290 live in lanes 0-15's slice
                a9[0] = (_Float16)ci;
                a9[1] = (_Float16)cj;
                a9[2] = (_Float16)(ci - cj);
            }
            A[tile][8] = a9;
        }

        v8f s0, s1;
        #pragma unroll
        for (int r = 0; r < 8; ++r) { s0[r] = 0.f; s1[r] = 0.f; }

        // Runtime n-tile loop: NOT unrolled, so the B-fragment LDS loads per
        // iteration cannot be hoisted out of the g-loop (that caused 720-VGPR
        // pressure and scratch spills when fully unrolled).
        #pragma unroll 1
        for (int t = 0; t < NT; ++t) {
            const int n = t * 16 + mloc;

            // Issue the small bias/W2 DS read FIRST: DS counter decrements in
            // issue order, so waiting on this early load does not imply waiting
            // on the B-fragment clause issued after it (previously the reverse
            // order forced an s_wait_dscnt 0 inside the WMMA chain).
            const float bias = sBias[n];
            const float w2   = sW2[n];

            const _Float16* wrow = &sW[n * KPAD + halfu * 16];

            // Preload ALL 9 B fragments into distinct registers so the
            // ds_loads form a clause and the scheduler overlaps later loads
            // with the first WMMAs via incremental s_wait_dscnt.
            v16h Bf[NC];
            #pragma unroll
            for (int c = 0; c < NC; ++c) {
                const int kc = (c < 8) ? c : 9;   // chunk 8 (cancer) folded into bias
                Bf[c] = *(const v16h*)(wrow + kc * 32);
            }

            v8f acc0 = {};
            v8f acc1 = {};
            #pragma unroll
            for (int c = 0; c < NC; ++c) {
                acc0 = __builtin_amdgcn_wmma_f32_16x16x32_f16(
                    false, A[0][c], false, Bf[c], (short)0, acc0, false, false);
                acc1 = __builtin_amdgcn_wmma_f32_16x16x32_f16(
                    false, A[1][c], false, Bf[c], (short)0, acc1, false, false);
            }

            // layer 2 folded in: sum_n relu(hid + b1') * W2
            #pragma unroll
            for (int r = 0; r < 8; ++r) {
                const float v0 = fmaxf(acc0[r] + bias, 0.f);
                const float v1 = fmaxf(acc1[r] + bias, 0.f);
                s0[r] += v0 * w2;
                s1[r] += v1 * w2;
            }
        }

        // reduce across the 16 lanes of each half (sum over N), then sigmoid + store
        #pragma unroll
        for (int r = 0; r < 8; ++r) {
            float a = s0[r];
            float b = s1[r];
            #pragma unroll
            for (int msk = 1; msk <= 8; msk <<= 1) {
                a += __shfl_xor(a, msk, 32);
                b += __shfl_xor(b, msk, 32);
            }
            if (mloc == r) {
                const int m0 = r + halfu * 8;      // lanes 0-15 -> edge r ; 16-31 -> edge 8+r
                out[ebase + m0]      = 1.f / (1.f + __expf(-(a + b2s)));
                out[ebase + 16 + m0] = 1.f / (1.f + __expf(-(b + b2s)));
            }
        }
    }
}

extern "C" void kernel_launch(void* const* d_in, const int* in_sizes, int n_in,
                              void* d_out, int out_size, void* d_ws, size_t ws_size,
                              hipStream_t stream) {
    const float*     h      = (const float*)d_in[0];
    const long long* ei     = (const long long*)d_in[1];
    const float*     cancer = (const float*)d_in[2];
    const float*     causal = (const float*)d_in[3];
    const float*     W1     = (const float*)d_in[4];
    const float*     b1     = (const float*)d_in[5];
    const float*     W2     = (const float*)d_in[6];
    const float*     b2     = (const float*)d_in[7];
    float* out = (float*)d_out;

    edge_mlp_wmma<<<dim3(GRID), dim3(THREADS), 0, stream>>>(
        h, ei, cancer, causal, W1, b1, W2, b2, out);
}
